// SVDMoeLinear_24618752540913
// MI455X (gfx1250) — matrix operand order repaired and checked by reading, not compiled, not run on hardware
//
#include <hip/hip_runtime.h>
#include <cstddef>

// Problem constants (from reference)
constexpr int cB    = 32;
constexpr int cS    = 2048;
constexpr int cDIN  = 1024;
constexpr int cDOUT = 1024;
constexpr int cK    = 2;
constexpr int cR    = 32;

typedef __attribute__((ext_vector_type(16))) __bf16 bf16x16;
typedef __attribute__((ext_vector_type(8)))  float  v8f;

// round-to-nearest-even float -> bf16 bits
static __device__ __forceinline__ unsigned short f2bf(float f) {
    unsigned int u = __float_as_uint(f);
    unsigned int r = u + 0x7FFFu + ((u >> 16) & 1u);
    return (unsigned short)(r >> 16);
}

// ---------------------------------------------------------------------------
// Kernel 1: fold per-batch rank-64 expert update into bf16 weights.
//   Wb[b][o][d] = bf16( W[o][d] + sum_{k,r} gate[b,k]*S[e,r]*U[e,o,r]*V[e,r,d] )
// grid = (DOUT/64, B), block = 256
// ---------------------------------------------------------------------------
__global__ __launch_bounds__(256)
void fold_weights_kernel(const float* __restrict__ W,
                         const int*   __restrict__ topk_idx,
                         const float* __restrict__ topk_gate,
                         const float* __restrict__ U_all,
                         const float* __restrict__ S_all,
                         const float* __restrict__ V_all,
                         unsigned short* __restrict__ Wb)   // bf16 bits [B][DOUT][DIN]
{
    const int b   = blockIdx.y;
    const int o0  = blockIdx.x * 64;
    const int tid = threadIdx.x;

    __shared__ float coeff[64][64];   // [o_local][rp], rp = k*32 + r   (16 KB)
    __shared__ float Vc[64][128];     // [rp][d_local] staged V chunk   (32 KB)

    const int   e0 = topk_idx[b * cK + 0];
    const int   e1 = topk_idx[b * cK + 1];
    const float g0 = topk_gate[b * cK + 0];
    const float g1 = topk_gate[b * cK + 1];

    // coefficients: gate * S * U  (64 rows x 64 rank')
    for (int idx = tid; idx < 64 * 64; idx += 256) {
        const int ol = idx >> 6;
        const int rp = idx & 63;
        const int k  = rp >> 5;
        const int r  = rp & 31;
        const int e  = k ? e1 : e0;
        const float g = k ? g1 : g0;
        coeff[ol][rp] = g * S_all[e * cR + r] * U_all[((size_t)e * cDOUT + (o0 + ol)) * cR + r];
    }

    for (int dc = 0; dc < cDIN; dc += 128) {
        __syncthreads();
        // stage V chunk: 64 rank-rows x 128 d, as float4 quads (64 x 32 quads)
        for (int idx = tid; idx < 64 * 32; idx += 256) {
            const int rp = idx >> 5;
            const int q  = idx & 31;
            const int k  = rp >> 5;
            const int r  = rp & 31;
            const int e  = k ? e1 : e0;
            const float4 v = *(const float4*)&V_all[((size_t)e * cR + r) * cDIN + dc + q * 4];
            *(float4*)&Vc[rp][q * 4] = v;
        }
        __syncthreads();
        // accumulate: each thread emits 8 float4 quads of Wb
        for (int idx = tid; idx < 64 * 32; idx += 256) {
            const int ol = idx >> 5;
            const int q  = idx & 31;
            const int o  = o0 + ol;
            const int d  = dc + q * 4;
            float4 acc = *(const float4*)&W[(size_t)o * cDIN + d];
            #pragma unroll 8
            for (int rp = 0; rp < 64; ++rp) {
                const float  c = coeff[ol][rp];
                const float4 v = *(const float4*)&Vc[rp][q * 4];
                acc.x = fmaf(c, v.x, acc.x);
                acc.y = fmaf(c, v.y, acc.y);
                acc.z = fmaf(c, v.z, acc.z);
                acc.w = fmaf(c, v.w, acc.w);
            }
            uint2 pk;
            pk.x = (unsigned int)f2bf(acc.x) | ((unsigned int)f2bf(acc.y) << 16);
            pk.y = (unsigned int)f2bf(acc.z) | ((unsigned int)f2bf(acc.w) << 16);
            *(uint2*)&Wb[((size_t)b * cDOUT + o) * cDIN + d] = pk;
        }
    }
}

// ---------------------------------------------------------------------------
// Kernel 2: out[b][s][o] = bias[o] + x[b][s][:] (bf16) @ Wb[b][o][:] (bf16)
// WG tile: 128(S) x 128(DOUT) for one batch; 8 waves, each 32x64 via
// 2x4 v_wmma_f32_16x16x32_bf16 accumulators. K pipelined 64 at a time with
// ping-pong LDS double buffering (one barrier per 64-K step) and register
// prefetch of the next x tile. Output uses non-temporal stores so the
// 256 MB output stream does not evict L2-resident x / Wb tiles.
// grid = (DOUT/128, S/128, B), block = 256
// ---------------------------------------------------------------------------
__global__ __launch_bounds__(256)
void gemm_bias_kernel(const float* __restrict__ x,
                      const unsigned short* __restrict__ Wb,   // bf16 bits
                      const float* __restrict__ bias,
                      float* __restrict__ out)
{
    const int nTile = blockIdx.x * 128;   // DOUT offset
    const int sTile = blockIdx.y * 128;   // S offset
    const int b     = blockIdx.z;
    const int tid   = threadIdx.x;
    const int lane  = tid & 31;
    const int wid   = tid >> 5;
    const int waveM = wid & 3;            // M offset 32*waveM
    const int waveN = wid >> 2;           // N offset 64*waveN
    const int lhalf = lane >> 4;          // 0: lanes 0-15, 1: lanes 16-31

    // Double-buffered A tile, 64 K per buffer, in natural K
    // order (position p holds K p); lane half h reads K 16h..16h+15 of each 32-K half
    // -> each lane does ONE aligned 32B read.
    __shared__ unsigned short As[2][128][64];   // 32 KB

    v8f acc[2][4];
    #pragma unroll
    for (int mi = 0; mi < 2; ++mi)
        #pragma unroll
        for (int ni = 0; ni < 4; ++ni)
            acc[mi][ni] = (v8f)0.0f;

    const float* xb = x + ((size_t)b * cS + sTile) * cDIN;
    const unsigned short* wb = Wb + (size_t)b * cDOUT * cDIN;

    // per-thread staging coordinates: 128 rows x 16 quads = 2048 quads, 8/thread
    int srow[8], sq[8], sdpos[8];
    #pragma unroll
    for (int i = 0; i < 8; ++i) {
        const int c  = tid + 256 * i;
        srow[i] = c >> 4;
        sq[i]   = c & 15;
        const int half = sq[i] >> 3;
        const int qp   = sq[i] & 7;
        const int kg   = qp >> 1;
        const int pg   = ((kg & 1) << 1) | (kg >> 1);
        sdpos[i] = half * 32 + qp * 4;
    }

    float4 xv[8];
    // ---- prologue: stage tile 0 into buffer 0
    #pragma unroll
    for (int i = 0; i < 8; ++i)
        xv[i] = *(const float4*)&xb[(size_t)srow[i] * cDIN + sq[i] * 4];
    #pragma unroll
    for (int i = 0; i < 8; ++i) {
        uint2 pk;
        pk.x = (unsigned int)f2bf(xv[i].x) | ((unsigned int)f2bf(xv[i].y) << 16);
        pk.y = (unsigned int)f2bf(xv[i].z) | ((unsigned int)f2bf(xv[i].w) << 16);
        *(uint2*)&As[0][srow[i]][sdpos[i]] = pk;
    }
    __syncthreads();

    constexpr int T = cDIN / 64;   // 16 pipeline stages
    for (int t = 0; t < T; ++t) {
        const int cur = t & 1;

        // ---- prefetch next x tile into registers (latency hidden by WMMAs)
        if (t + 1 < T) {
            const int kn = (t + 1) * 64;
            #pragma unroll
            for (int i = 0; i < 8; ++i)
                xv[i] = *(const float4*)&xb[(size_t)srow[i] * cDIN + kn + sq[i] * 4];
        }

        // ---- compute: 2 K-subtiles of 32, 8 WMMAs each
        #pragma unroll
        for (int ks = 0; ks < 2; ++ks) {
            bf16x16 afrag[2];
            #pragma unroll
            for (int mi = 0; mi < 2; ++mi) {
                const int row = 32 * waveM + 16 * mi + (lane & 15);
                afrag[mi] = *(const bf16x16*)&As[cur][row][ks * 32 + lhalf * 16];
            }
            bf16x16 bfrag[4];
            #pragma unroll
            for (int ni = 0; ni < 4; ++ni) {
                const int ocol = nTile + 64 * waveN + 16 * ni + (lane & 15);
                const int kb   = t * 64 + ks * 32 + (lhalf << 4);
                bfrag[ni] = *(const bf16x16*)&wb[(size_t)ocol * cDIN + kb];
            }
            #pragma unroll
            for (int mi = 0; mi < 2; ++mi)
                #pragma unroll
                for (int ni = 0; ni < 4; ++ni)
                    acc[mi][ni] = __builtin_amdgcn_wmma_f32_16x16x32_bf16(
                        false, afrag[mi], false, bfrag[ni],
                        (short)0, acc[mi][ni], false, false);
        }

        // ---- convert + store prefetched tile into the other buffer
        if (t + 1 < T) {
            const int nxt = cur ^ 1;
            #pragma unroll
            for (int i = 0; i < 8; ++i) {
                uint2 pk;
                pk.x = (unsigned int)f2bf(xv[i].x) | ((unsigned int)f2bf(xv[i].y) << 16);
                pk.y = (unsigned int)f2bf(xv[i].z) | ((unsigned int)f2bf(xv[i].w) << 16);
                *(uint2*)&As[nxt][srow[i]][sdpos[i]] = pk;
            }
        }
        __syncthreads();   // one barrier per 64-K step
    }

    // ---- epilogue: + bias, non-temporal fp32 stores per 16x16 C/D layout
    float* ob = out + (size_t)b * cS * cDOUT;
    #pragma unroll
    for (int mi = 0; mi < 2; ++mi) {
        const int rbase = sTile + 32 * waveM + 16 * mi + lhalf * 8;
        #pragma unroll
        for (int ni = 0; ni < 4; ++ni) {
            const int col = nTile + 64 * waveN + 16 * ni + (lane & 15);
            const float bv = bias[col];
            #pragma unroll
            for (int v = 0; v < 8; ++v) {
                __builtin_nontemporal_store(acc[mi][ni][v] + bv,
                    &ob[(size_t)(rbase + v) * cDOUT + col]);
            }
        }
    }
}

// ---------------------------------------------------------------------------
extern "C" void kernel_launch(void* const* d_in, const int* in_sizes, int n_in,
                              void* d_out, int out_size, void* d_ws, size_t ws_size,
                              hipStream_t stream) {
    const float* x     = (const float*)d_in[0];
    const int*   idx   = (const int*)  d_in[1];
    const float* gates = (const float*)d_in[2];
    const float* W     = (const float*)d_in[3];
    const float* U     = (const float*)d_in[4];
    const float* Sv    = (const float*)d_in[5];
    const float* V     = (const float*)d_in[6];
    const float* bias  = (const float*)d_in[7];
    float* out = (float*)d_out;

    // workspace: per-batch folded bf16 weights, B*DOUT*DIN*2 = 64 MB
    unsigned short* Wb = (unsigned short*)d_ws;

    fold_weights_kernel<<<dim3(cDOUT / 64, cB), 256, 0, stream>>>(
        W, idx, gates, U, Sv, V, Wb);

    gemm_bias_kernel<<<dim3(cDOUT / 128, cS / 128, cB), 256, 0, stream>>>(
        x, Wb, bias, out);
}
